// TrainGNNModel_17678085390367
// MI455X (gfx1250) — matrix operand order repaired and checked by reading, not compiled, not run on hardware
//
#include <hip/hip_runtime.h>
#include <hip/hip_bf16.h>

typedef __attribute__((ext_vector_type(16))) __bf16 v16bf;
typedef __attribute__((ext_vector_type(8)))  __bf16 v8bf;
typedef __attribute__((ext_vector_type(8)))  float  v8f;

#define IN_DIM 31
#define HID    128
#define NCLS   20

__device__ __forceinline__ __bf16 f2bf(float f) { return (__bf16)f; }

// ---------------- weight pre-swizzle into WMMA B-fragment layout ----------------
// B fragment (16-bit, 32x16 K-major tile): lane = (K-half<<4) | column,
// per-lane 16 contiguous halves = K values kbase..kbase+15 for that column.
// Packed index: (((jt*kblocks + kb)*32 + lane)*16 + i)
__global__ void k_pack_weights(const float* __restrict__ W, __bf16* __restrict__ Bp,
                               int K, int Kpad, int Nout, int ntiles) {
    int t = blockIdx.x * blockDim.x + threadIdx.x;
    int kblocks = Kpad >> 5;
    int total = ntiles * kblocks * 32;
    if (t >= total) return;
    int lane = t & 31;
    int kb   = (t >> 5) % kblocks;
    int jt   = (t >> 5) / kblocks;
    int col   = jt * 16 + (lane & 15);
    int kbase = kb * 32 + (lane >> 4) * 16;
    __bf16* out = Bp + (size_t)t * 16;
#pragma unroll
    for (int i = 0; i < 16; ++i) {
        int k = kbase + i;
        float v = (k < K && col < Nout) ? W[(size_t)k * Nout + col] : 0.0f;
        out[i] = f2bf(v);
    }
}

// ---------------- x (f32, K=31) -> bf16 padded to K=32 ----------------
__global__ void k_convert_x(const float* __restrict__ x, __bf16* __restrict__ xb, int N) {
    int idx = blockIdx.x * blockDim.x + threadIdx.x;
    if (idx >= N * 32) return;
    int node = idx >> 5, k = idx & 31;
    float v = (k < IN_DIM) ? x[(size_t)node * IN_DIM + k] : 0.0f;
    xb[idx] = f2bf(v);
}

// ---------------- degree / dinv ----------------
__global__ void k_init_deg(float* __restrict__ deg, int N) {
    int i = blockIdx.x * blockDim.x + threadIdx.x;
    if (i < N) deg[i] = 1.0f;                    // self-loop
}
__global__ void k_edge_deg(const int* __restrict__ dst, float* __restrict__ deg, int E) {
    int e = blockIdx.x * blockDim.x + threadIdx.x;
    if (e < E) atomicAdd(&deg[dst[e]], 1.0f);
}
__global__ void k_rsqrt(float* __restrict__ deg, int N) {
    int i = blockIdx.x * blockDim.x + threadIdx.x;
    if (i < N) deg[i] = rsqrtf(deg[i]);          // deg >= 1 always
}

__global__ void k_zero(float* __restrict__ p, int n) {
    int i = blockIdx.x * blockDim.x + threadIdx.x;
    if (i < n) p[i] = 0.0f;
}

// ---------------- WMMA GEMM: C[M,Nout] = A[M,K](bf16) * Bp(packed bf16) ----------------
// Block = ntiles waves; wave jt computes the 16x16 tile (blockIdx.x, jt).
// A fragment (contiguous K order, matching packed B): lane<16 holds K 0-15,
// lane>=16 holds K 16-31, row = lane&15.
__global__ void k_wmma_gemm(const __bf16* __restrict__ A, int lda,
                            const __bf16* __restrict__ Bp, int kblocks,
                            const float* __restrict__ bias,
                            float* __restrict__ outF, __bf16* __restrict__ outH,
                            int ldo, int nvalid, int relu, int M) {
    const int lane = threadIdx.x & 31;
    const int jt   = threadIdx.x >> 5;
    const int hi   = lane >> 4;
    const int n    = lane & 15;

    int rowA = blockIdx.x * 16 + n;
    if (rowA > M - 1) rowA = M - 1;              // branchless clamp, EXEC stays full

    const __bf16* aptr = A + (size_t)rowA * lda + hi * 16;
    const __bf16* bptr = Bp + ((size_t)jt * kblocks * 32 + lane) * 16;

    v8f acc = {0.f, 0.f, 0.f, 0.f, 0.f, 0.f, 0.f, 0.f};
    for (int kb = 0; kb < kblocks; ++kb) {
        v8bf a0 = *(const v8bf*)(aptr);
        v8bf a1 = *(const v8bf*)(aptr + 8);
        v8bf b0 = *(const v8bf*)(bptr);
        v8bf b1 = *(const v8bf*)(bptr + 8);
        v16bf a = __builtin_shufflevector(a0, a1, 0,1,2,3,4,5,6,7,8,9,10,11,12,13,14,15);
        v16bf b = __builtin_shufflevector(b0, b1, 0,1,2,3,4,5,6,7,8,9,10,11,12,13,14,15);
        acc = __builtin_amdgcn_wmma_f32_16x16x32_bf16(
                  false, a, false, b, (short)0, acc, false, false);
        aptr += 32;
        bptr += 512;                              // 32 lanes * 16 halves
    }

    const int col = jt * 16 + n;
    float bv = (bias != nullptr && col < nvalid) ? bias[col] : 0.0f;
#pragma unroll
    for (int v = 0; v < 8; ++v) {
        int row = blockIdx.x * 16 + hi * 8 + v;   // C layout: M = hi*8 + vgpr
        if (row < M && col < nvalid) {
            float val = acc[v] + bv;
            if (relu) val = fmaxf(val, 0.0f);
            size_t o = (size_t)row * ldo + col;
            if (outF) outF[o] = val;
            if (outH) outH[o] = f2bf(val);
        }
    }
}

// ---------------- edge scatter: agg[dst] += xw[src] * dinv[src]*dinv[dst] ----------------
// One wave per edge; each lane moves a float4 slice of the 128-float row.
__global__ void k_edge_aggregate(const int* __restrict__ esrc, const int* __restrict__ edst,
                                 const float* __restrict__ dinv, const float* __restrict__ xw,
                                 float* __restrict__ agg, int E) {
    int e = blockIdx.x * (blockDim.x >> 5) + (threadIdx.x >> 5);
    if (e >= E) return;
    int lane = threadIdx.x & 31;
    int s = esrc[e];
    int d = edst[e];
    float nrm = dinv[s] * dinv[d];
    const float4 v = ((const float4*)(xw + (size_t)s * HID))[lane];
    float* a = agg + (size_t)d * HID + lane * 4;
    atomicAdd(a + 0, v.x * nrm);                  // non-returning -> global_atomic_add_f32
    atomicAdd(a + 1, v.y * nrm);
    atomicAdd(a + 2, v.z * nrm);
    atomicAdd(a + 3, v.w * nrm);
}

// ---------------- epilogue: h = relu(agg + xw*dinv^2 + b), store bf16 ----------------
__global__ void k_gcn_finish(const float* __restrict__ agg, const float* __restrict__ xw,
                             const float* __restrict__ dinv, const float* __restrict__ bias,
                             __bf16* __restrict__ hb, int N) {
    int idx = blockIdx.x * blockDim.x + threadIdx.x;
    if (idx >= N * HID) return;
    int node = idx >> 7;
    int col  = idx & (HID - 1);
    float di = dinv[node];
    float v = agg[idx] + xw[idx] * di * di + bias[col];
    hb[idx] = f2bf(fmaxf(v, 0.0f));
}

extern "C" void kernel_launch(void* const* d_in, const int* in_sizes, int n_in,
                              void* d_out, int out_size, void* d_ws, size_t ws_size,
                              hipStream_t stream) {
    const float* x     = (const float*)d_in[0];
    const int*   ei    = (const int*)d_in[1];
    const float* W_enc = (const float*)d_in[2];
    const float* b_enc = (const float*)d_in[3];
    const float* W1    = (const float*)d_in[4];
    const float* b1    = (const float*)d_in[5];
    const float* W2    = (const float*)d_in[6];
    const float* b2    = (const float*)d_in[7];
    const float* W_cls = (const float*)d_in[8];
    const float* b_cls = (const float*)d_in[9];
    float* out = (float*)d_out;
    (void)n_in; (void)out_size; (void)ws_size;

    const int N = in_sizes[0] / IN_DIM;
    const int E = in_sizes[1] / 2;
    const int* esrc = ei;
    const int* edst = ei + E;

    // workspace carve-out (256B aligned)
    char* ws = (char*)d_ws;
    size_t off = 0;
    auto take = [&](size_t bytes) -> void* {
        off = (off + 255) & ~(size_t)255;
        void* p = ws + off;
        off += bytes;
        return p;
    };
    float*  dinv  = (float*) take((size_t)N * 4);
    __bf16* xb    = (__bf16*)take((size_t)N * 32 * 2);
    __bf16* hb    = (__bf16*)take((size_t)N * HID * 2);
    float*  xw    = (float*) take((size_t)N * HID * 4);
    float*  agg   = (float*) take((size_t)N * HID * 4);
    __bf16* Wencp = (__bf16*)take((size_t)8 * 1 * 32 * 16 * 2);
    __bf16* W1p   = (__bf16*)take((size_t)8 * 4 * 32 * 16 * 2);
    __bf16* W2p   = (__bf16*)take((size_t)8 * 4 * 32 * 16 * 2);
    __bf16* Wclsp = (__bf16*)take((size_t)2 * 4 * 32 * 16 * 2);

    const int mtiles = (N + 15) / 16;            // N=100000 -> 6250, exact
    const int nh = N * HID;

    // weight pre-swizzle
    k_pack_weights<<<(8*1*32 + 255)/256, 256, 0, stream>>>(W_enc, Wencp, IN_DIM, 32,  HID,  8);
    k_pack_weights<<<(8*4*32 + 255)/256, 256, 0, stream>>>(W1,    W1p,   HID,    HID, HID,  8);
    k_pack_weights<<<(8*4*32 + 255)/256, 256, 0, stream>>>(W2,    W2p,   HID,    HID, HID,  8);
    k_pack_weights<<<(2*4*32 + 255)/256, 256, 0, stream>>>(W_cls, Wclsp, HID,    HID, NCLS, 2);

    // x -> bf16 (K padded to 32)
    k_convert_x<<<(N*32 + 255)/256, 256, 0, stream>>>(x, xb, N);

    // symmetric-norm coefficients
    k_init_deg<<<(N + 255)/256, 256, 0, stream>>>(dinv, N);
    k_edge_deg<<<(E + 255)/256, 256, 0, stream>>>(edst, dinv, E);
    k_rsqrt  <<<(N + 255)/256, 256, 0, stream>>>(dinv, N);

    // encoder: h = relu(x @ W_enc + b_enc), bf16 out
    k_wmma_gemm<<<mtiles, 8*32, 0, stream>>>(xb, 32, Wencp, 1, b_enc,
                                             nullptr, hb, HID, HID, 1, N);

    // ---- GCN layer 1 ----
    k_wmma_gemm<<<mtiles, 8*32, 0, stream>>>(hb, HID, W1p, 4, nullptr,
                                             xw, nullptr, HID, HID, 0, N);
    k_zero<<<(nh + 255)/256, 256, 0, stream>>>(agg, nh);
    k_edge_aggregate<<<(E + 7)/8, 256, 0, stream>>>(esrc, edst, dinv, xw, agg, E);
    k_gcn_finish<<<(nh + 255)/256, 256, 0, stream>>>(agg, xw, dinv, b1, hb, N);

    // ---- GCN layer 2 ----
    k_wmma_gemm<<<mtiles, 8*32, 0, stream>>>(hb, HID, W2p, 4, nullptr,
                                             xw, nullptr, HID, HID, 0, N);
    k_zero<<<(nh + 255)/256, 256, 0, stream>>>(agg, nh);
    k_edge_aggregate<<<(E + 7)/8, 256, 0, stream>>>(esrc, edst, dinv, xw, agg, E);
    k_gcn_finish<<<(nh + 255)/256, 256, 0, stream>>>(agg, xw, dinv, b2, hb, N);

    // classifier: out = h @ W_cls + b_cls (f32, 20 valid cols of 32)
    k_wmma_gemm<<<mtiles, 2*32, 0, stream>>>(hb, HID, Wclsp, 4, b_cls,
                                             out, nullptr, NCLS, NCLS, 0, N);
}
